// Attention_10969346474829
// MI455X (gfx1250) — matrix operand order repaired and checked by reading, not compiled, not run on hardware
//
#include <hip/hip_runtime.h>
#include <stdint.h>

#define DIMSZ  2048
#define HEADS  32
#define DHEAD  64
#define DINNER 2048
#define BATCH  2
#define SEQ    2048

typedef __attribute__((ext_vector_type(16))) __bf16 v16bf;
typedef __attribute__((ext_vector_type(8)))  float  v8f;
typedef __attribute__((ext_vector_type(8)))  unsigned int v8u;
typedef __attribute__((ext_vector_type(4)))  unsigned int u32x4;
typedef __attribute__((ext_vector_type(8)))  unsigned int u32x8;

union FragB { v8u u; v16bf b; };

static __device__ inline v8f v8f_zero() {
  v8f z = {0.f,0.f,0.f,0.f,0.f,0.f,0.f,0.f};
  return z;
}

static __device__ inline v8f wmma_bf16(v16bf a, v16bf b, v8f c) {
  // D = A(16x32 bf16) * B(32x16 bf16) + C(16x16 f32)
  return __builtin_amdgcn_wmma_f32_16x16x32_bf16(false, a, false, b, (short)0, c, false, false);
}

static __device__ inline unsigned short f2bf(float f) {
  unsigned int u = __float_as_uint(f);
  u += 0x7FFFu + ((u >> 16) & 1u);       // round-to-nearest-even
  return (unsigned short)(u >> 16);
}
static __device__ inline float bf2f(unsigned short h) {
  return __uint_as_float(((unsigned int)h) << 16);
}

// Low 32 bits of a generic pointer to a __shared__ object are the LDS byte
// offset within the workgroup allocation (ISA 10.2 aperture mapping).
static __device__ inline unsigned lds_off(const void* p) {
  return (unsigned)(unsigned long long)p;
}

// Per-lane async copy of 16B global -> LDS (ASYNCcnt-tracked).
static __device__ inline void async_g2l_b128(unsigned ldsaddr, unsigned long long gaddr) {
  asm volatile("global_load_async_to_lds_b128 %0, %1, off" :: "v"(ldsaddr), "v"(gaddr) : "memory");
}
static __device__ inline void wait_asynccnt0() {
  asm volatile("s_wait_asynccnt 0" ::: "memory");
}

// TDM: 2D tile load global -> LDS. tile = tile1 rows x (tile0*4) bytes,
// global row stride = stride0 (4-byte units). Issued once per executing wave.
static __device__ inline void tdm_load_2d(unsigned ldsaddr, unsigned long long gaddr,
                                          unsigned dim0, unsigned dim1,
                                          unsigned tile0, unsigned tile1,
                                          unsigned long long stride0) {
  u32x4 g0;
  g0[0] = 1u;                                   // count=1, user mode
  g0[1] = ldsaddr;                              // lds_addr [63:32]
  g0[2] = (unsigned)gaddr;                      // global_addr [95:64]
  g0[3] = (unsigned)(gaddr >> 32) | 0x80000000u; // addr[56:32] + type=2 [127:126]
  u32x8 g1;
  g1[0] = 2u << 16;                             // data_size = 2 (4 bytes)
  g1[1] = (dim0 & 0xFFFFu) << 16;               // tensor_dim0 [79:48]
  g1[2] = ((dim0 >> 16) & 0xFFFFu) | ((dim1 & 0xFFFFu) << 16);   // + tensor_dim1 [111:80]
  g1[3] = ((dim1 >> 16) & 0xFFFFu) | ((tile0 & 0xFFFFu) << 16);  // + tile_dim0 [127:112]
  g1[4] = (tile1 & 0xFFFFu);                    // tile_dim1 [143:128], tile_dim2=0
  g1[5] = (unsigned)stride0;                    // tensor_dim0_stride [207:160]
  g1[6] = (unsigned)((stride0 >> 32) & 0xFFFFu);
  g1[7] = 0u;                                   // tensor_dim1_stride unused (2D)
  asm volatile("tensor_load_to_lds %0, %1" :: "s"(g0), "s"(g1) : "memory");
}

// ---------------------------------------------------------------- convert
__global__ void f32_to_bf16(const float* __restrict__ src,
                            unsigned short* __restrict__ dst, int n) {
  int i = blockIdx.x * blockDim.x + threadIdx.x;
  int stride = gridDim.x * blockDim.x;
  for (; i < n; i += stride) dst[i] = f2bf(src[i]);
}

// ---------------------------------------------------------------- rmsnorm
__global__ __launch_bounds__(256) void rmsnorm_kernel(
    const float* __restrict__ x, const float* __restrict__ gamma,
    unsigned short* __restrict__ xn) {
  __shared__ float red[256];
  const int row = blockIdx.x;
  const int t = threadIdx.x;
  const float* xr = x + (long)row * DIMSZ;
  float s = 0.f;
  for (int i = 0; i < DIMSZ / 256; ++i) { float v = xr[t + i * 256]; s += v * v; }
  red[t] = s;
  __syncthreads();
  for (int off = 128; off > 0; off >>= 1) {
    if (t < off) red[t] += red[t + off];
    __syncthreads();
  }
  const float scale = 45.254834f / fmaxf(sqrtf(red[0]), 1e-12f); // sqrt(2048)/||x||
  for (int i = 0; i < DIMSZ / 256; ++i) {
    int c = t + i * 256;
    xn[(long)row * DIMSZ + c] = f2bf(xr[c] * scale * gamma[c]);
  }
}

// ---------------------------------------------------------------- GEMM
// C[M,N] = A[M,K](bf16) * B[K,N](bf16), fp32 accum.
// Block tile 64x128, 8 waves of 32x32, BK=32. A tile staged by the TDM.
template <bool OUT_F32>
__global__ __launch_bounds__(256) void gemm_bf16(
    const unsigned short* __restrict__ A, const unsigned short* __restrict__ Bw,
    void* __restrict__ C, int M, int N, int K) {
  __shared__ unsigned int sA[64 * 16];   // [m][kpair] (64B rows, TDM-packed)
  __shared__ unsigned int sB[128 * 16];  // transposed: [n][kpair]
  const int tid  = threadIdx.x;
  const int lane = tid & 31, wave = tid >> 5;
  const int half = lane >> 4, ln = lane & 15;
  const int m0 = blockIdx.y * 64;
  const int n0 = blockIdx.x * 128;
  const int mi = wave >> 2, ni = wave & 3;   // 2x4 wave grid
  const int Kp = K >> 1;

  v8f acc[2][2];
  for (int a = 0; a < 2; ++a)
    for (int b = 0; b < 2; ++b) acc[a][b] = v8f_zero();

  for (int k0 = 0; k0 < K; k0 += 32) {
    if (wave == 0) {                               // TDM: A tile 64 rows x 64B
      unsigned long long ga = (unsigned long long)A + ((long)m0 * K + k0) * 2;
      tdm_load_2d(lds_off(&sA[0]), ga, (unsigned)Kp, 64u, 16u, 64u,
                  (unsigned long long)Kp);
    }
    for (int i = 0; i < 8; ++i) {                  // stage B transposed 128x32
      int e = i * 256 + tid;
      int n = e & 127, kp = e >> 7;
      unsigned int lo = Bw[(long)(k0 + 2 * kp) * N + n0 + n];
      unsigned int hi = Bw[(long)(k0 + 2 * kp + 1) * N + n0 + n];
      sB[n * 16 + kp] = lo | (hi << 16);
    }
    if (k0 + 32 < K)                               // prefetch next B tile
      __builtin_prefetch(&Bw[(long)(k0 + 32) * N + n0 + (tid & 127)], 0, 1);
    if (wave == 0) __builtin_amdgcn_s_wait_tensorcnt(0);
    __syncthreads();

    FragB af[2], bfr[2];
    for (int ms = 0; ms < 2; ++ms) {
      int row = mi * 32 + ms * 16 + ln;
      for (int v = 0; v < 8; ++v) {
        int kp = half * 8 + v;                           // contiguous k order (matches B)
        af[ms].u[v] = sA[row * 16 + kp];
      }
    }
    for (int ns = 0; ns < 2; ++ns) {
      int col = ni * 32 + ns * 16 + ln;
      for (int v = 0; v < 8; ++v)
        bfr[ns].u[v] = sB[col * 16 + half * 8 + v];      // contiguous k order (matches A)
    }
    for (int ms = 0; ms < 2; ++ms)
      for (int ns = 0; ns < 2; ++ns)
        acc[ms][ns] = wmma_bf16(af[ms].b, bfr[ns].b, acc[ms][ns]);
    __syncthreads();
  }

  for (int ms = 0; ms < 2; ++ms)
    for (int ns = 0; ns < 2; ++ns)
      for (int r = 0; r < 8; ++r) {
        long row = m0 + mi * 32 + ms * 16 + r + 8 * half;  // C-layout M
        long col = n0 + ni * 32 + ns * 16 + ln;            // C-layout N
        float val = acc[ms][ns][r];
        if (OUT_F32) ((float*)C)[row * N + col] = val;
        else         ((unsigned short*)C)[row * N + col] = f2bf(val);
      }
}

// ---------------------------------------------------------------- rotary + head split
__global__ void rotary_split(const unsigned short* __restrict__ qlin,
                             const unsigned short* __restrict__ kvlin,
                             const float* __restrict__ rot,
                             unsigned short* __restrict__ qr,
                             unsigned short* __restrict__ kr,
                             unsigned short* __restrict__ vr) {
  const long total = (long)BATCH * SEQ * DINNER;
  long i = (long)blockIdx.x * blockDim.x + threadIdx.x;
  const long stride = (long)gridDim.x * blockDim.x;
  for (; i < total; i += stride) {
    int c  = (int)(i % DINNER);
    int bn = (int)(i / DINNER);
    int h = c >> 6, d = c & 63;
    int b = bn / SEQ, n = bn % SEQ;
    float ang = rot[n * DHEAD + d];
    float cs = cosf(ang), sn = sinf(ang);
    int   dj  = (d < 32) ? d + 32 : d - 32;
    float sgn = (d < 32) ? -1.f : 1.f;
    float tq = bf2f(qlin[(long)bn * DINNER + c]);
    float rq = sgn * bf2f(qlin[(long)bn * DINNER + h * 64 + dj]);
    float tk = bf2f(kvlin[(long)bn * 2 * DINNER + c]);
    float rk = sgn * bf2f(kvlin[(long)bn * 2 * DINNER + h * 64 + dj]);
    float tv = bf2f(kvlin[(long)bn * 2 * DINNER + DINNER + c]);
    long o = (((long)(b * HEADS + h)) * SEQ + n) * DHEAD + d;
    qr[o] = f2bf(tq * cs + rq * sn);
    kr[o] = f2bf(tk * cs + rk * sn);
    vr[o] = f2bf(tv);
  }
}

// ---------------------------------------------------------------- flash attention
// grid: (SEQ/64, BATCH*HEADS), 128 threads (4 waves, 16 query rows/wave).
__global__ __launch_bounds__(128) void flash_attn(
    const unsigned short* __restrict__ qr, const unsigned short* __restrict__ kr,
    const unsigned short* __restrict__ vr, unsigned short* __restrict__ ao) {
  __shared__ unsigned int sK[32 * 32];    // [j][dpair] (4KB, contiguous in global)
  __shared__ unsigned int sVt[64 * 16];   // transposed: [d][jpair]
  __shared__ unsigned int sP[4][16 * 16]; // per-wave P tile, [m][jpair]

  const int tid = threadIdx.x;
  const int wave = tid >> 5, lane = tid & 31;
  const int half = lane >> 4, ln = lane & 15;
  const int qtile = blockIdx.x;
  const int bh = blockIdx.y;
  const int b = bh / HEADS, h = bh % HEADS;
  const int qbase = qtile * 64 + wave * 16;

  const unsigned int* q32 = (const unsigned int*)qr;

  FragB qf[2];                                    // Q rows, d-chunks 0-31 / 32-63
  {
    long qrow = (long)bh * SEQ + qbase + ln;
    for (int c = 0; c < 2; ++c)
      for (int v = 0; v < 8; ++v) {
        int kp = c * 16 + half * 8 + v;
        qf[c].u[v] = q32[qrow * 32 + kp];
      }
  }

  float rowmax[8], rowsum[8];
  v8f accO[4];
  for (int r = 0; r < 8; ++r) { rowmax[r] = -3.0e38f; rowsum[r] = 0.f; }
  for (int dt = 0; dt < 4; ++dt) accO[dt] = v8f_zero();

  const unsigned long long kblk_bytes = (unsigned long long)kr + (unsigned long long)bh * SEQ * DHEAD * 2;
  const unsigned sK_base = lds_off(&sK[0]);

  const int jend = (qtile + 1) * 64;              // causal bound for this block
  for (int j0 = 0; j0 < jend; j0 += 32) {
    // async copy of contiguous 4KB K block (rows j0..j0+31) into LDS
    for (int i = 0; i < 2; ++i) {
      int e = i * 128 + tid;                      // 256 x 16B = 4KB
      async_g2l_b128(sK_base + e * 16, kblk_bytes + (unsigned long long)j0 * 128 + e * 16);
    }
    if (j0 + 32 < jend)                           // prefetch next K block
      __builtin_prefetch((const void*)(kblk_bytes + (unsigned long long)(j0 + 32) * 128 + tid * 16), 0, 1);
    for (int i = 0; i < 8; ++i) {                 // stage V transposed 64x32
      int e = i * 128 + tid;
      int d = e >> 4, jp = e & 15;
      unsigned int lo = vr[((long)bh * SEQ + j0 + 2 * jp) * DHEAD + d];
      unsigned int hi = vr[((long)bh * SEQ + j0 + 2 * jp + 1) * DHEAD + d];
      sVt[e] = lo | (hi << 16);
    }
    wait_asynccnt0();
    __syncthreads();

    float sv[2][8];
    for (int sub = 0; sub < 2; ++sub) {           // S = Q * K^T (16x16 subtiles)
      FragB bf0, bf1;
      for (int v = 0; v < 8; ++v) {
        bf0.u[v] = sK[(sub * 16 + ln) * 32 +      half * 8 + v];
        bf1.u[v] = sK[(sub * 16 + ln) * 32 + 16 + half * 8 + v];
      }
      v8f s = v8f_zero();
      s = wmma_bf16(qf[0].b, bf0.b, s);
      s = wmma_bf16(qf[1].b, bf1.b, s);
      int jg = j0 + sub * 16 + ln;
      for (int r = 0; r < 8; ++r) {
        int mg = qbase + r + 8 * half;
        sv[sub][r] = (jg > mg) ? -3.0e38f : s[r] * 0.125f;  // scale = 1/sqrt(64)
      }
    }

    float alpha[8];
    for (int r = 0; r < 8; ++r) {                 // online softmax per row
      float bm = fmaxf(sv[0][r], sv[1][r]);
      for (int m = 8; m >= 1; m >>= 1) bm = fmaxf(bm, __shfl_xor(bm, m, 32));
      float nm = fmaxf(rowmax[r], bm);
      float a  = __expf(rowmax[r] - nm);
      float p0 = __expf(sv[0][r] - nm);
      float p1 = __expf(sv[1][r] - nm);
      sv[0][r] = p0; sv[1][r] = p1;
      float ps = p0 + p1;
      for (int m = 8; m >= 1; m >>= 1) ps += __shfl_xor(ps, m, 32);
      rowsum[r] = rowsum[r] * a + ps;
      rowmax[r] = nm;
      alpha[r]  = a;
    }
    for (int dt = 0; dt < 4; ++dt)
      for (int r = 0; r < 8; ++r) accO[dt][r] *= alpha[r];

    // C-layout -> A-layout for P via per-wave LDS
    unsigned short* sPh = (unsigned short*)sP[wave];
    for (int sub = 0; sub < 2; ++sub)
      for (int r = 0; r < 8; ++r)
        sPh[(r + 8 * half) * 32 + sub * 16 + ln] = f2bf(sv[sub][r]);
    asm volatile("s_wait_dscnt 0" ::: "memory");  // same-wave cross-lane RAW via LDS

    FragB pf;
    for (int v = 0; v < 8; ++v) {
      int kp = half * 8 + v;
      pf.u[v] = sP[wave][ln * 16 + kp];
    }
    for (int dt = 0; dt < 4; ++dt) {              // O += P * V
      FragB vf;
      for (int v = 0; v < 8; ++v)
        vf.u[v] = sVt[(dt * 16 + ln) * 16 + half * 8 + v];
      accO[dt] = wmma_bf16(pf.b, vf.b, accO[dt]);
    }
    __syncthreads();
  }

  for (int dt = 0; dt < 4; ++dt)
    for (int r = 0; r < 8; ++r) {
      int mg = qbase + r + 8 * half;
      int col = h * DHEAD + dt * 16 + ln;
      float val = accO[dt][r] / fmaxf(rowsum[r], 1e-20f);
      ao[((long)(b * SEQ + mg)) * DINNER + col] = f2bf(val);
    }
}

// ---------------------------------------------------------------- launcher
extern "C" void kernel_launch(void* const* d_in, const int* in_sizes, int n_in,
                              void* d_out, int out_size, void* d_ws, size_t ws_size,
                              hipStream_t stream) {
  (void)in_sizes; (void)n_in; (void)out_size; (void)ws_size;
  const float* x     = (const float*)d_in[0];
  const float* rot   = (const float*)d_in[1];
  const float* gamma = (const float*)d_in[2];
  const float* Wq    = (const float*)d_in[3];
  const float* Wkv   = (const float*)d_in[4];
  const float* Wout  = (const float*)d_in[5];

  char* ws = (char*)d_ws;
  const size_t SZ_WQ    = (size_t)DIMSZ * DINNER * 2;
  const size_t SZ_WKV   = (size_t)DIMSZ * 2 * DINNER * 2;
  const size_t SZ_WOUT  = (size_t)DINNER * DIMSZ * 2;
  const size_t SZ_XN    = (size_t)BATCH * SEQ * DIMSZ * 2;
  const size_t SZ_QLIN  = (size_t)BATCH * SEQ * DINNER * 2;
  const size_t SZ_KVLIN = (size_t)BATCH * SEQ * 2 * DINNER * 2;
  const size_t SZ_HD    = (size_t)BATCH * HEADS * SEQ * DHEAD * 2;

  unsigned short* wq_b   = (unsigned short*)(ws);
  unsigned short* wkv_b  = (unsigned short*)(ws + SZ_WQ);
  unsigned short* wout_b = (unsigned short*)(ws + SZ_WQ + SZ_WKV);
  unsigned short* xn     = (unsigned short*)(ws + SZ_WQ + SZ_WKV + SZ_WOUT);
  unsigned short* qlin   = (unsigned short*)((char*)xn + SZ_XN);
  unsigned short* kvlin  = (unsigned short*)((char*)qlin + SZ_QLIN);
  unsigned short* q_r    = (unsigned short*)((char*)kvlin + SZ_KVLIN);
  unsigned short* k_r    = (unsigned short*)((char*)q_r + SZ_HD);
  unsigned short* v_r    = (unsigned short*)((char*)k_r + SZ_HD);
  unsigned short* ao     = xn;  // xn dead after QKV GEMMs -> reuse for attn out

  f32_to_bf16<<<4096, 256, 0, stream>>>(Wq, wq_b, DIMSZ * DINNER);
  f32_to_bf16<<<4096, 256, 0, stream>>>(Wkv, wkv_b, DIMSZ * 2 * DINNER);
  f32_to_bf16<<<4096, 256, 0, stream>>>(Wout, wout_b, DINNER * DIMSZ);

  rmsnorm_kernel<<<BATCH * SEQ, 256, 0, stream>>>(x, gamma, xn);

  dim3 gq(DINNER / 128, (BATCH * SEQ) / 64);
  gemm_bf16<false><<<gq, 256, 0, stream>>>(xn, wq_b, qlin, BATCH * SEQ, DINNER, DIMSZ);
  dim3 gkv((2 * DINNER) / 128, (BATCH * SEQ) / 64);
  gemm_bf16<false><<<gkv, 256, 0, stream>>>(xn, wkv_b, kvlin, BATCH * SEQ, 2 * DINNER, DIMSZ);

  rotary_split<<<8192, 256, 0, stream>>>(qlin, kvlin, rot, q_r, k_r, v_r);

  dim3 gfa(SEQ / 64, BATCH * HEADS);
  flash_attn<<<gfa, 128, 0, stream>>>(q_r, k_r, v_r, ao);

  dim3 go(DIMSZ / 128, (BATCH * SEQ) / 64);
  gemm_bf16<true><<<go, 256, 0, stream>>>(ao, wout_b, (float*)d_out, BATCH * SEQ, DIMSZ, DINNER);
}
